// HyperLoopedGPT_9629316678120
// MI455X (gfx1250) — matrix operand order repaired and checked by reading, not compiled, not run on hardware
//
#include <hip/hip_runtime.h>
#include <hip/hip_bf16.h>
#include <math.h>

// ---------------- problem constants (from reference) ----------------
#define B_     2
#define T_     2048
#define D_     512
#define NH_    8
#define HD_    64          // D/NH
#define V_     16384
#define DFF_   2048        // 4*D
#define NLOOP_ 6
#define FREQ_  256
#define M_     (B_*T_)     // 4096 token rows

typedef __attribute__((ext_vector_type(16))) __bf16        v16bf;
typedef __attribute__((ext_vector_type(8)))  float         v8f;
typedef __attribute__((ext_vector_type(4)))  unsigned int  v4u;
typedef __attribute__((ext_vector_type(8)))  int           v8i;
typedef __attribute__((ext_vector_type(4)))  int           v4i;

#if defined(__HIP_DEVICE_COMPILE__) && __has_builtin(__builtin_amdgcn_tensor_load_to_lds)
#define HAVE_TDM 1
#else
#define HAVE_TDM 0
#endif

// ---------------- helpers ----------------
__device__ __forceinline__ unsigned short f2bf(float f) {
  unsigned int u = __float_as_uint(f);
  unsigned int r = u + 0x7FFFu + ((u >> 16) & 1u);   // round-to-nearest-even
  return (unsigned short)(r >> 16);
}

// A fragment (16x32 bf16). Lane holds row m = lane&15.
// element i -> k = (i>>3)*16 + (lane>>4)*8 + (i&7)  (ISA 7.12.2 16-bit A table)
__device__ __forceinline__ v16bf load_fragA(const unsigned short* rowptr, int half) {
  v16bf f;
  uint4* p = reinterpret_cast<uint4*>(&f);
  p[0] = *reinterpret_cast<const uint4*>(rowptr + half * 8);
  p[1] = *reinterpret_cast<const uint4*>(rowptr + 16 + half * 8);
  return f;
}

// B fragment (32x16 bf16). Lane holds col n = lane&15.
// element i -> k = (i>>3)*16 + (lane>>4)*8 + (i&7)  (same order as A)
// rowptr points at Bt[n][k0] (K-contiguous, i.e. column-major B).
__device__ __forceinline__ v16bf load_fragB(const unsigned short* rowptr, int half) {
  v16bf f;
  uint4* p = reinterpret_cast<uint4*>(&f);
  p[0] = *reinterpret_cast<const uint4*>(rowptr + half * 8);
  p[1] = *reinterpret_cast<const uint4*>(rowptr + 16 + half * 8);
  return f;
}

__device__ __forceinline__ v8f wmma_bf16(v16bf a, v16bf b, v8f c) {
  return __builtin_amdgcn_wmma_f32_16x16x32_bf16(false, a, false, b, (short)0, c, false, false);
}

__device__ __forceinline__ float silu_f(float x) { return x / (1.0f + __expf(-x)); }

#if HAVE_TDM
// Issue one TDM 2D tile load: tile_rows x tile_cols bf16 elements, row stride
// row_stride_elts, packed contiguously into LDS at lds_off.
// D# packing per CDNA5 ISA ch.8: group0 {count=1,type=2,lds,global}, group1
// {data_size=2B, tensor_dim0=stride, tensor_dim1=big, tile_dim0/1, dim0_stride}.
__device__ __forceinline__ void tdm_load_tile_bf16(unsigned lds_off,
                                                   const unsigned short* gsrc,
                                                   unsigned tile_cols,
                                                   unsigned tile_rows,
                                                   unsigned row_stride_elts) {
  unsigned long long ga = (unsigned long long)(uintptr_t)gsrc;
  v4u g0;
  g0[0] = 1u;                                              // count=1, user mode
  g0[1] = lds_off;                                         // lds_addr (bytes)
  g0[2] = (unsigned)ga;                                    // global_addr[31:0]
  g0[3] = (unsigned)((ga >> 32) & 0x1FFFFFFull) | (2u << 30); // addr[56:32] | type=2
  unsigned long long td0 = row_stride_elts;                // tensor row length
  unsigned long long st0 = row_stride_elts;                // dim0 stride (elements)
  unsigned long long q0 = (1ull << 16)                     // data_size = 2 bytes
                        | ((td0 & 0xFFFFull) << 48);       // tensor_dim0[15:0]
  unsigned long long q1 = ((td0 >> 16) & 0xFFFFull)        // tensor_dim0[31:16]
                        | (0x100000ull << 16)              // tensor_dim1 (large)
                        | ((unsigned long long)tile_cols << 48); // tile_dim0
  unsigned long long q2 = (unsigned long long)tile_rows    // tile_dim1
                        | ((st0 & 0xFFFFFFFFull) << 32);   // dim0_stride[31:0]
  unsigned long long q3 = (st0 >> 32) & 0xFFFFull;         // dim0_stride[47:32]
  v8i g1;
  g1[0] = (int)q0; g1[1] = (int)(q0 >> 32);
  g1[2] = (int)q1; g1[3] = (int)(q1 >> 32);
  g1[4] = (int)q2; g1[5] = (int)(q2 >> 32);
  g1[6] = (int)q3; g1[7] = (int)(q3 >> 32);
  v4i z = {0, 0, 0, 0};
#if defined(__clang_major__) && (__clang_major__ >= 23)
  v8i z8 = {0, 0, 0, 0, 0, 0, 0, 0};
  __builtin_amdgcn_tensor_load_to_lds(g0, g1, z, z, z8, 0);
#else
  __builtin_amdgcn_tensor_load_to_lds(g0, g1, z, z, 0);
#endif
}
#endif  // HAVE_TDM

// ---------------- weight convert + transpose: W[K][N] f32 -> Wt[N][K] bf16 ----------------
__global__ __launch_bounds__(256) void wcvt_kernel(const float* __restrict__ W,
                                                   unsigned short* __restrict__ Wt,
                                                   int K, int N) {
  size_t i = (size_t)blockIdx.x * 256 + threadIdx.x;
  if (i >= (size_t)K * N) return;
  int k = (int)(i / N), n = (int)(i % N);
  Wt[(size_t)n * K + k] = f2bf(W[i]);
}

// ---------------- timestep embedding + adaLN modulation, all 6 steps ----------------
__global__ __launch_bounds__(256) void tstep_mod_kernel(
    const float* __restrict__ te_w1, const float* __restrict__ te_b1,
    const float* __restrict__ te_w2, const float* __restrict__ te_b2,
    const float* __restrict__ ada_w, const float* __restrict__ ada_b,
    float* __restrict__ mod_all) {
  __shared__ float emb[FREQ_];
  __shared__ float h1[D_];
  __shared__ float st[D_];
  const int step = blockIdx.x;
  const int t = threadIdx.x;
  const float fs = (float)step;
  {
    const int half = FREQ_ / 2;
    float fr = __expf(-logf(10000.0f) * (float)(t % half) / (float)half);
    emb[t] = (t < half) ? cosf(fs * fr) : sinf(fs * fr);
  }
  __syncthreads();
  for (int d = t; d < D_; d += 256) {
    float a = te_b1[d];
    for (int i = 0; i < FREQ_; ++i) a += emb[i] * te_w1[i * D_ + d];
    h1[d] = silu_f(a);
  }
  __syncthreads();
  for (int d = t; d < D_; d += 256) {
    float a = te_b2[d];
    for (int j = 0; j < D_; ++j) a += h1[j] * te_w2[j * D_ + d];
    st[d] = silu_f(a);                      // silu(t_emb)
  }
  __syncthreads();
  for (int o = t; o < 4 * D_; o += 256) {
    float a = ada_b[o];
    for (int d = 0; d < D_; ++d) a += st[d] * ada_w[d * (4 * D_) + o];
    mod_all[step * (4 * D_) + o] = a;
  }
}

// ---------------- embedding gather + LayerNorm -> x (f32) ----------------
__global__ __launch_bounds__(256) void embed_ln_kernel(
    const int* __restrict__ idx, const float* __restrict__ tok,
    const float* __restrict__ g, const float* __restrict__ bb,
    float* __restrict__ x) {
  __shared__ float r1[256];
  __shared__ float r2[256];
  const int m = blockIdx.x, t = threadIdx.x;
  const float* e = tok + (size_t)idx[m] * D_;
  float v0 = e[t], v1 = e[t + 256];
  r1[t] = v0 + v1;  r2[t] = v0 * v0 + v1 * v1;
  __syncthreads();
  for (int s = 128; s > 0; s >>= 1) {
    if (t < s) { r1[t] += r1[t + s]; r2[t] += r2[t + s]; }
    __syncthreads();
  }
  float mu = r1[0] * (1.0f / D_);
  float var = r2[0] * (1.0f / D_) - mu * mu;
  float r = rsqrtf(var + 1e-5f);
  float* xr = x + (size_t)m * D_;
  xr[t]       = (v0 - mu) * r * g[t] + bb[t];
  xr[t + 256] = (v1 - mu) * r * g[t + 256] + bb[t + 256];
}

// ---------------- RMSNorm * (1+scale) -> bf16 activations ----------------
__global__ __launch_bounds__(256) void rmsmod_kernel(
    const float* __restrict__ x, const float* __restrict__ scale,
    unsigned short* __restrict__ xin) {
  __shared__ float r1[256];
  const int m = blockIdx.x, t = threadIdx.x;
  const float* xr = x + (size_t)m * D_;
  float v0 = xr[t], v1 = xr[t + 256];
  r1[t] = v0 * v0 + v1 * v1;
  __syncthreads();
  for (int s = 128; s > 0; s >>= 1) {
    if (t < s) r1[t] += r1[t + s];
    __syncthreads();
  }
  float r = rsqrtf(r1[0] * (1.0f / D_) + 1.1920929e-7f);   // FLT_EPSILON
  unsigned short* o = xin + (size_t)m * D_;
  o[t]       = f2bf(v0 * r * (1.0f + scale[t]));
  o[t + 256] = f2bf(v1 * r * (1.0f + scale[t + 256]));
}

// ---------------- final LayerNorm -> bf16 ----------------
__global__ __launch_bounds__(256) void lnf_kernel(
    const float* __restrict__ x, const float* __restrict__ g,
    const float* __restrict__ bb, unsigned short* __restrict__ xin) {
  __shared__ float r1[256];
  __shared__ float r2[256];
  const int m = blockIdx.x, t = threadIdx.x;
  const float* xr = x + (size_t)m * D_;
  float v0 = xr[t], v1 = xr[t + 256];
  r1[t] = v0 + v1;  r2[t] = v0 * v0 + v1 * v1;
  __syncthreads();
  for (int s = 128; s > 0; s >>= 1) {
    if (t < s) { r1[t] += r1[t + s]; r2[t] += r2[t + s]; }
    __syncthreads();
  }
  float mu = r1[0] * (1.0f / D_);
  float var = r2[0] * (1.0f / D_) - mu * mu;
  float r = rsqrtf(var + 1e-5f);
  unsigned short* o = xin + (size_t)m * D_;
  o[t]       = f2bf((v0 - mu) * r * g[t] + bb[t]);
  o[t + 256] = f2bf((v1 - mu) * r * g[t + 256] + bb[t + 256]);
}

// ---------------- generic WMMA GEMM: C = A[M,K] * Bt[N,K]^T + bias ----------------
// Block tile 128M x 128N, 8 waves each 32M x 64N (2x4 fragments, 8 WMMA per K-step).
// LDS tiles double-buffered and filled by the Tensor Data Mover (async, TENSORcnt).
#define EPI_QKV   0
#define EPI_SILU  1
#define EPI_RESID 2
#define EPI_OUT   3

template <int EPI>
__global__ __launch_bounds__(256) void gemm_wmma(
    const unsigned short* __restrict__ A,   // [M][K] bf16
    const unsigned short* __restrict__ Bt,  // [N][K] bf16 (pre-transposed weight)
    const float* __restrict__ bias,         // [N]
    int N, int K,
    float* __restrict__ outf,               // EPI_RESID: x (accum), EPI_OUT: fp32 out
    unsigned short* __restrict__ outb,      // EPI_SILU: h (bf16)
    const float* __restrict__ gate,         // EPI_RESID
    unsigned short* __restrict__ qb,
    unsigned short* __restrict__ kb,
    unsigned short* __restrict__ vtb) {
#if HAVE_TDM
  __shared__ __attribute__((aligned(16))) unsigned short As[2][128 * 32];
  __shared__ __attribute__((aligned(16))) unsigned short Bs[2][128 * 32];
#else
  __shared__ __attribute__((aligned(16))) unsigned short As[1][128 * 32];
  __shared__ __attribute__((aligned(16))) unsigned short Bs[1][128 * 32];
#endif
  const int nBlkN = N >> 7;
  const int bm = (blockIdx.x / nBlkN) * 128;
  const int bn = (blockIdx.x % nBlkN) * 128;
  const int tid = threadIdx.x;
  const int w = tid >> 5, lane = tid & 31;
  const int half = lane >> 4, l16 = lane & 15;
  const int wm = (w & 3) * 32, wn = (w >> 2) * 64;

  v8f acc[2][4] = {};
  int buf = 0;
#if HAVE_TDM
  if (w == 0) {
    tdm_load_tile_bf16((unsigned)(uintptr_t)&As[0][0], A + (size_t)bm * K, 32, 128, K);
    tdm_load_tile_bf16((unsigned)(uintptr_t)&Bs[0][0], Bt + (size_t)bn * K, 32, 128, K);
  }
#endif
  for (int k0 = 0; k0 < K; k0 += 32) {
#if HAVE_TDM
    if (w == 0) {
      if (k0 + 32 < K) {
        tdm_load_tile_bf16((unsigned)(uintptr_t)&As[buf ^ 1][0],
                           A + (size_t)bm * K + (k0 + 32), 32, 128, K);
        tdm_load_tile_bf16((unsigned)(uintptr_t)&Bs[buf ^ 1][0],
                           Bt + (size_t)bn * K + (k0 + 32), 32, 128, K);
        __builtin_amdgcn_s_wait_tensorcnt(2);   // oldest 2 (current bufs) done
      } else {
        __builtin_amdgcn_s_wait_tensorcnt(0);
      }
    }
#else
    // manual staging fallback: 128x32 bf16 tiles, 512 uint4 each
#pragma unroll
    for (int r = 0; r < 2; ++r) {
      int i = tid + r * 256;
      int row = i >> 2, off = i & 3;
      reinterpret_cast<uint4*>(&As[0][0])[i] =
          *reinterpret_cast<const uint4*>(A + (size_t)(bm + row) * K + k0 + off * 8);
      reinterpret_cast<uint4*>(&Bs[0][0])[i] =
          *reinterpret_cast<const uint4*>(Bt + (size_t)(bn + row) * K + k0 + off * 8);
    }
    if (k0 + 32 < K)
      __builtin_prefetch(A + (size_t)(bm + (tid >> 1)) * K + k0 + 32, 0, 3);
#endif
    __syncthreads();
    const unsigned short* pa = &As[buf][0];
    const unsigned short* pb = &Bs[buf][0];
    v16bf a0 = load_fragA(pa + (wm + l16) * 32, half);
    v16bf a1 = load_fragA(pa + (wm + 16 + l16) * 32, half);
#pragma unroll
    for (int nt = 0; nt < 4; ++nt) {
      v16bf bfr = load_fragB(pb + (wn + nt * 16 + l16) * 32, half);
      acc[0][nt] = wmma_bf16(a0, bfr, acc[0][nt]);
      acc[1][nt] = wmma_bf16(a1, bfr, acc[1][nt]);
    }
    __syncthreads();
#if HAVE_TDM
    buf ^= 1;
#endif
  }
  // epilogue: C layout -> row = mt*16 + v + 8*half, col = nt*16 + l16
#pragma unroll
  for (int mt = 0; mt < 2; ++mt)
#pragma unroll
    for (int nt = 0; nt < 4; ++nt)
#pragma unroll
      for (int v = 0; v < 8; ++v) {
        int row = bm + wm + mt * 16 + v + 8 * half;
        int col = bn + wn + nt * 16 + l16;
        float val = acc[mt][nt][v] + bias[col];
        if (EPI == EPI_QKV) {
          int part = col >> 9;             // 0:q 1:k 2:v
          int d = col & 511;
          int h = d >> 6, hd = d & 63;
          int b = row >> 11, t = row & (T_ - 1);
          size_t bh = (size_t)(b * NH_ + h);
          if (part == 0)      qb[(bh * T_ + t) * HD_ + hd] = f2bf(val * 0.125f); // fold 1/sqrt(HD)
          else if (part == 1) kb[(bh * T_ + t) * HD_ + hd] = f2bf(val);
          else                vtb[(bh * HD_ + hd) * T_ + t] = f2bf(val);          // v transposed
        } else if (EPI == EPI_SILU) {
          outb[(size_t)row * N + col] = f2bf(silu_f(val));
        } else if (EPI == EPI_RESID) {
          outf[(size_t)row * N + col] += gate[col] * val;
        } else {
          outf[(size_t)row * N + col] = val;
        }
      }
}

// ---------------- flash attention (transposed S = K*Q^T), ALiBi + causal ----------------
__global__ __launch_bounds__(256) void attn_kernel(
    const unsigned short* __restrict__ qb,   // [B,NH,T,HD] bf16, pre-scaled
    const unsigned short* __restrict__ kb,   // [B,NH,T,HD] bf16
    const unsigned short* __restrict__ vtb,  // [B,NH,HD,T] bf16
    unsigned short* __restrict__ y) {        // [B,T,D] bf16
  __shared__ __attribute__((aligned(16))) unsigned short Ks[32 * 64];
  __shared__ __attribute__((aligned(16))) unsigned short Vs[64 * 32];
  __shared__ __attribute__((aligned(16))) unsigned short Ps[8][16 * 40]; // per-wave, padded rows
  const int nqb = T_ / 128;
  const int qb0 = (blockIdx.x % nqb) * 128;
  const int bh = blockIdx.x / nqb;
  const int b = bh / NH_, h = bh % NH_;
  const int tid = threadIdx.x, w = tid >> 5, lane = tid & 31;
  const int half = lane >> 4, l16 = lane & 15;
  const int qw = qb0 + w * 16;             // wave's q-tile base
  const int q = qw + l16;                  // this lane's query column
  const float slope = exp2f(-(float)(h + 1));  // 2^(-8/NH*(h+1)), NH=8

  // loop-invariant Q B-fragments (n=q across lanes, k=hd)
  const unsigned short* qrow = qb + ((size_t)bh * T_ + q) * HD_;
  v16bf fq0 = load_fragB(qrow, half);
  v16bf fq1 = load_fragB(qrow + 32, half);

  v8f o[4] = {};                           // O^T frags: hd tiles x (n=q)
  float m_run = -__builtin_inff(), l_run = 0.0f;

  const unsigned short* kbase = kb + (size_t)bh * T_ * HD_;
  const unsigned short* vbase = vtb + (size_t)bh * HD_ * T_;
  const int nst = qb0 / 32 + 4;            // causal: keys up to qb0+127
  for (int st = 0; st < nst; ++st) {
    const int s0 = st * 32;
    // K tile [32 s][64 hd] : contiguous 4KB
    reinterpret_cast<uint4*>(Ks)[tid] =
        *reinterpret_cast<const uint4*>(kbase + (size_t)s0 * HD_ + tid * 8);
    // V^T tile [64 hd][32 s]
    {
      int r = tid >> 2, off = tid & 3;
      reinterpret_cast<uint4*>(Vs)[tid] =
          *reinterpret_cast<const uint4*>(vbase + (size_t)r * T_ + s0 + off * 8);
    }
    __syncthreads();
    if (s0 <= qw + 15) {                   // wave-uniform skip of fully-masked tiles
      // S^T fragments: M=s (2 tiles of 16), N=q ; K=hd in 2 chunks
      v8f sf[2] = {};
#pragma unroll
      for (int mt = 0; mt < 2; ++mt) {
        v16bf a0 = load_fragA(Ks + (mt * 16 + l16) * 64, half);
        v16bf a1 = load_fragA(Ks + (mt * 16 + l16) * 64 + 32, half);
        sf[mt] = wmma_bf16(a0, fq0, sf[mt]);
        sf[mt] = wmma_bf16(a1, fq1, sf[mt]);
      }
      float pv[2][8];
      float mloc = -__builtin_inff();
#pragma unroll
      for (int mt = 0; mt < 2; ++mt)
#pragma unroll
        for (int v = 0; v < 8; ++v) {
          int s = s0 + mt * 16 + v + 8 * half;
          float val = (s <= q) ? (sf[mt][v] - slope * (float)(q - s)) : -__builtin_inff();
          pv[mt][v] = val;
          mloc = fmaxf(mloc, val);
        }
      float mtile = fmaxf(mloc, __shfl_xor(mloc, 16, 32));
      float mnew = fmaxf(m_run, mtile);
      float corr = __expf(m_run - mnew);
      float lsum = 0.0f;
#pragma unroll
      for (int mt = 0; mt < 2; ++mt)
#pragma unroll
        for (int v = 0; v < 8; ++v) {
          int s = s0 + mt * 16 + v + 8 * half;
          float p = (s <= q) ? __expf(pv[mt][v] - mnew) : 0.0f;
          pv[mt][v] = p;
          lsum += p;
        }
      lsum += __shfl_xor(lsum, 16, 32);
      l_run = l_run * corr + lsum;
      m_run = mnew;
#pragma unroll
      for (int i = 0; i < 4; ++i)
#pragma unroll
        for (int v = 0; v < 8; ++v) o[i][v] *= corr;
      // rotate P (C-layout of S^T) into B-layout via per-wave LDS [q=16][s=32], row stride 40
      unsigned short* pw = &Ps[w][0];
#pragma unroll
      for (int mt = 0; mt < 2; ++mt)
#pragma unroll
        for (int v = 0; v < 8; ++v) {
          int sl = mt * 16 + v + 8 * half;
          pw[l16 * 40 + sl] = f2bf(pv[mt][v]);
        }
      asm volatile("s_wait_dscnt 0x0" ::: "memory");   // wave-local LDS RAW ordering
      v16bf fp = load_fragB(pw + l16 * 40, half);      // n=q, k=s
#pragma unroll
      for (int i = 0; i < 4; ++i) {
        v16bf av = load_fragA(Vs + (i * 16 + l16) * 32, half); // m=hd, k=s
        o[i] = wmma_bf16(av, fp, o[i]);
      }
    }
    __syncthreads();
  }
  const float inv = 1.0f / l_run;
  unsigned short* yrow = y + ((size_t)(b * T_ + q)) * D_ + h * HD_;
#pragma unroll
  for (int i = 0; i < 4; ++i)
#pragma unroll
    for (int v = 0; v < 8; ++v) {
      int hd = i * 16 + v + 8 * half;
      yrow[hd] = f2bf(o[i][v] * inv);
    }
}

// ---------------- host-side orchestration ----------------
extern "C" void kernel_launch(void* const* d_in, const int* in_sizes, int n_in,
                              void* d_out, int out_size, void* d_ws, size_t ws_size,
                              hipStream_t stream) {
  (void)in_sizes; (void)n_in; (void)out_size; (void)ws_size;
  const int*   idx      = (const int*)  d_in[0];
  const float* tok      = (const float*)d_in[1];
  const float* emb_g    = (const float*)d_in[2];
  const float* emb_b    = (const float*)d_in[3];
  const float* c_attn_w = (const float*)d_in[4];
  const float* c_attn_b = (const float*)d_in[5];
  const float* c_proj_w = (const float*)d_in[6];
  const float* c_proj_b = (const float*)d_in[7];
  const float* mlp_w1   = (const float*)d_in[8];
  const float* mlp_b1   = (const float*)d_in[9];
  const float* mlp_w2   = (const float*)d_in[10];
  const float* mlp_b2   = (const float*)d_in[11];
  const float* ada_w    = (const float*)d_in[12];
  const float* ada_b    = (const float*)d_in[13];
  const float* te_w1    = (const float*)d_in[14];
  const float* te_b1    = (const float*)d_in[15];
  const float* te_w2    = (const float*)d_in[16];
  const float* te_b2    = (const float*)d_in[17];
  const float* lnf_g    = (const float*)d_in[18];
  const float* lnf_b    = (const float*)d_in[19];
  const float* head_w   = (const float*)d_in[20];
  const float* head_b   = (const float*)d_in[21];
  float* out = (float*)d_out;

  // workspace carve-up (256B aligned slices)
  char* ws = (char*)d_ws;
  size_t off = 0;
  auto carve = [&](size_t bytes) -> void* {
    void* p = ws + off;
    off += (bytes + 255) & ~(size_t)255;
    return p;
  };
  float*          x       = (float*)         carve((size_t)M_ * D_ * 4);
  unsigned short* xin     = (unsigned short*)carve((size_t)M_ * D_ * 2);
  unsigned short* qbuf    = (unsigned short*)carve((size_t)M_ * D_ * 2);
  unsigned short* kbuf    = (unsigned short*)carve((size_t)M_ * D_ * 2);
  unsigned short* vtbuf   = (unsigned short*)carve((size_t)M_ * D_ * 2);
  unsigned short* ybuf    = (unsigned short*)carve((size_t)M_ * D_ * 2);
  unsigned short* hbuf    = (unsigned short*)carve((size_t)M_ * DFF_ * 2);
  unsigned short* wqkv_t  = (unsigned short*)carve((size_t)D_ * 3 * D_ * 2);
  unsigned short* wproj_t = (unsigned short*)carve((size_t)D_ * D_ * 2);
  unsigned short* w1_t    = (unsigned short*)carve((size_t)D_ * DFF_ * 2);
  unsigned short* w2_t    = (unsigned short*)carve((size_t)DFF_ * D_ * 2);
  unsigned short* whead_t = (unsigned short*)carve((size_t)D_ * V_ * 2);
  float*          mod_all = (float*)         carve((size_t)NLOOP_ * 4 * D_ * 4);

  auto cdiv = [](size_t a, size_t b) { return (int)((a + b - 1) / b); };

  // one-time (per launch) weight convert+transpose into bf16 [N][K]
  wcvt_kernel<<<cdiv((size_t)D_ * 3 * D_, 256), 256, 0, stream>>>(c_attn_w, wqkv_t, D_, 3 * D_);
  wcvt_kernel<<<cdiv((size_t)D_ * D_, 256),     256, 0, stream>>>(c_proj_w, wproj_t, D_, D_);
  wcvt_kernel<<<cdiv((size_t)D_ * DFF_, 256),   256, 0, stream>>>(mlp_w1,   w1_t,   D_, DFF_);
  wcvt_kernel<<<cdiv((size_t)DFF_ * D_, 256),   256, 0, stream>>>(mlp_w2,   w2_t,   DFF_, D_);
  wcvt_kernel<<<cdiv((size_t)D_ * V_, 256),     256, 0, stream>>>(head_w,   whead_t, D_, V_);

  tstep_mod_kernel<<<NLOOP_, 256, 0, stream>>>(te_w1, te_b1, te_w2, te_b2, ada_w, ada_b, mod_all);
  embed_ln_kernel<<<M_, 256, 0, stream>>>(idx, tok, emb_g, emb_b, x);

  for (int step = 0; step < NLOOP_; ++step) {
    const float* modv = mod_all + (size_t)step * 4 * D_;
    const float* scale_msa = modv;
    const float* scale_mlp = modv + D_;
    const float* gate_msa  = modv + 2 * D_;
    const float* gate_mlp  = modv + 3 * D_;

    rmsmod_kernel<<<M_, 256, 0, stream>>>(x, scale_msa, xin);
    gemm_wmma<EPI_QKV><<<(M_ / 128) * (3 * D_ / 128), 256, 0, stream>>>(
        xin, wqkv_t, c_attn_b, 3 * D_, D_, nullptr, nullptr, nullptr, qbuf, kbuf, vtbuf);
    attn_kernel<<<B_ * NH_ * (T_ / 128), 256, 0, stream>>>(qbuf, kbuf, vtbuf, ybuf);
    gemm_wmma<EPI_RESID><<<(M_ / 128) * (D_ / 128), 256, 0, stream>>>(
        ybuf, wproj_t, c_proj_b, D_, D_, x, nullptr, gate_msa, nullptr, nullptr, nullptr);

    rmsmod_kernel<<<M_, 256, 0, stream>>>(x, scale_mlp, xin);
    gemm_wmma<EPI_SILU><<<(M_ / 128) * (DFF_ / 128), 256, 0, stream>>>(
        xin, w1_t, mlp_b1, DFF_, D_, nullptr, hbuf, nullptr, nullptr, nullptr, nullptr);
    gemm_wmma<EPI_RESID><<<(M_ / 128) * (D_ / 128), 256, 0, stream>>>(
        hbuf, w2_t, mlp_b2, D_, DFF_, x, nullptr, gate_mlp, nullptr, nullptr, nullptr);
  }

  lnf_kernel<<<M_, 256, 0, stream>>>(x, lnf_g, lnf_b, xin);
  gemm_wmma<EPI_OUT><<<(M_ / 128) * (V_ / 128), 256, 0, stream>>>(
      xin, whead_t, head_b, V_, D_, out, nullptr, nullptr, nullptr, nullptr, nullptr);
}
